// PGAConjugateLinear_14130442404106
// MI455X (gfx1250) — matrix operand order repaired and checked by reading, not compiled, not run on hardware
//
#include <hip/hip_runtime.h>
#include <stdint.h>
#include <stddef.h>

typedef __attribute__((ext_vector_type(16))) _Float16 v16h;
typedef __attribute__((ext_vector_type(8)))  _Float16 v8h;
typedef __attribute__((ext_vector_type(4)))  _Float16 v4h;
typedef __attribute__((ext_vector_type(8)))  float    v8f;
typedef __attribute__((ext_vector_type(4)))  float    v4f;

// ---------------- Cl(3,0,1) blade-mask product sign -----------------------
// Blades as 4-bit masks (bit g = generator e_g). e0^2 = 0; e1,e2,e3 ^2 = 1.
// sign = (-1)^{#transpositions to interleave a after b}, 0 if e0 repeats.
__device__ __forceinline__ int pga_mul_sign(int a, int b) {
  if (a & b & 1) return 0;             // e0 contracted with metric 0
  int cnt = 0;
  int t = a >> 1;
  while (t) { cnt += __popc(t & b); t >>= 1; }
  return (cnt & 1) ? -1 : 1;
}

// ---------------- 1) cast x (f32) -> A (f16), vectorized ------------------
__global__ void __launch_bounds__(256)
pga_xcast_kernel(const float* __restrict__ src, _Float16* __restrict__ dst, int n4) {
  int t = blockIdx.x * blockDim.x + threadIdx.x;
  if (t >= n4) return;
  v4f v = *(const v4f*)(src + (size_t)t * 4);
  v4h h;
  h[0] = (_Float16)v[0]; h[1] = (_Float16)v[1];
  h[2] = (_Float16)v[2]; h[3] = (_Float16)v[3];
  *(v4h*)(dst + (size_t)t * 4) = h;
}

// ---------------- 2) build effective weight Bt (N,K) + bias contribs ------
// T[p][r] = sum_q sign(q,p) * krev[q^p] * sign(q^r, q) * k[q^r]
// Bt[(o*3+p)*K + (i*3+r)] = w[o,i] * T[p][r]          (r = 0..2)
// biasC[i*N + (o*3+p)]    = w[o,i] * T[p][3] * e0[i]  (e123 column, (I,N) layout)
__global__ void __launch_bounds__(256)
pga_build_weff_kernel(const float* __restrict__ weight,   // (O,I)
                      const float* __restrict__ action,   // (O,I,8)
                      const float* __restrict__ e0,       // (I)
                      _Float16* __restrict__ Bt,          // (N=3O, K=3I)
                      float* __restrict__ biasC,          // (I, N)
                      int O, int I) {
  const int t = blockIdx.x * blockDim.x + threadIdx.x;
  if (t >= O * I) return;
  const int o = t / I;
  const int i = t - o * I;
  const int K = 3 * I;
  const int N = 3 * O;

  float a[8];
#pragma unroll
  for (int j = 0; j < 8; ++j) a[j] = action[(size_t)t * 8 + j];

  // ACTION_BLADES (0,5,6,7,8,9,10,15) -> masks (0,3,5,9,6,10,12,15)
  float k[16];
#pragma unroll
  for (int m = 0; m < 16; ++m) k[m] = 0.0f;
  k[0] = a[0];  k[3] = a[1];  k[5]  = a[2];  k[9]  = a[3];
  k[6] = a[4];  k[10] = a[5]; k[12] = a[6];  k[15] = a[7];

  float kr[16];   // reverse: grade 2,3 negate
#pragma unroll
  for (int m = 0; m < 16; ++m) {
    int g = __popc(m);
    kr[m] = (((g * (g - 1)) >> 1) & 1) ? -k[m] : k[m];
  }

  const float w = weight[t];
  const int pmask[3] = {7, 11, 13};       // e012, e013, e023
  const int rmask[4] = {7, 11, 13, 14};   // + e123 (homogeneous)

#pragma unroll
  for (int pi = 0; pi < 3; ++pi) {
#pragma unroll
    for (int ri = 0; ri < 4; ++ri) {
      const int p = pmask[pi], r = rmask[ri];
      float s = 0.0f;
#pragma unroll
      for (int q = 0; q < 16; ++q) {       // fully constant-folds
        const int s1 = pga_mul_sign(q, p);
        if (s1 == 0) continue;
        const int s2 = pga_mul_sign(q ^ r, q);
        if (s2 == 0) continue;
        const float term = kr[q ^ p] * k[q ^ r];
        s = (s1 * s2 > 0) ? (s + term) : (s - term);
      }
      if (ri < 3) {
        Bt[(size_t)(o * 3 + pi) * K + (size_t)(i * 3 + ri)] = (_Float16)(w * s);
      } else {
        biasC[(size_t)i * N + (o * 3 + pi)] = w * s * e0[i];
      }
    }
  }
}

// ---------------- 3) deterministic bias reduction over I (coalesced) -------
__global__ void __launch_bounds__(256)
pga_bias_reduce_kernel(const float* __restrict__ biasC, float* __restrict__ bias,
                       int N, int I) {
  int n = blockIdx.x * blockDim.x + threadIdx.x;
  if (n >= N) return;
  float s = 0.0f;
  for (int i = 0; i < I; ++i) s += biasC[(size_t)i * N + n];   // lanes contiguous
  bias[n] = s;
}

// ---------------- 4) WMMA GEMM: C(M,N) = A(M,K) * Bt(N,K)^T + bias ---------
// Block: 256 threads = 8 waves (4 M x 2 N); each wave 32x64 via 2x4 WMMA
// fragments -> block tile 128x128, K-step 32. 8 WMMA : 12 b128 per step.
// M=4096, N=1536, K=1536 divide the tiling exactly.
__global__ void __launch_bounds__(256)
pga_gemm_kernel(const _Float16* __restrict__ A,   // (M,K) row-major f16
                const _Float16* __restrict__ Bt,  // (N,K) row-major f16
                const float* __restrict__ bias,   // (N)
                float* __restrict__ C,            // (M,N) row-major f32
                int M, int N, int K) {
  const int lane = threadIdx.x & 31;
  const int wave = threadIdx.x >> 5;
  const int l15  = lane & 15;
  const int hi   = lane >> 4;          // lane half: 0 or 1

  const int mWave = blockIdx.x * 128 + (wave & 3) * 32;
  const int nWave = blockIdx.y * 128 + (wave >> 2) * 64;

  v8f acc[2][4] = {};

  const _Float16* aRow[2];
#pragma unroll
  for (int mi = 0; mi < 2; ++mi)
    aRow[mi] = A + (size_t)(mWave + mi * 16 + l15) * K;
  const _Float16* bRow[4];
#pragma unroll
  for (int ni = 0; ni < 4; ++ni)
    bRow[ni] = Bt + (size_t)(nWave + ni * 16 + l15) * K;

  // A 16x32 f16 frag: lane<16 -> K {0..7,16..23}; lane>=16 -> {8..15,24..31}
  const int aOff = hi * 8;
  // B 32x16 f16 frag: lane<16 -> K {0..7,16..23}; lane>=16 -> {8..15,24..31}
  const int bOff = hi * 8;

  for (int k0 = 0; k0 < K; k0 += 32) {
    if (k0 + 32 < K) {                       // prefetch next K tile
      __builtin_prefetch(aRow[0] + k0 + 32, 0, 3);
      __builtin_prefetch(bRow[0] + k0 + 32, 0, 3);
    }

    v16h af[2];
#pragma unroll
    for (int mi = 0; mi < 2; ++mi) {
      v8h lo = *(const v8h*)(aRow[mi] + k0 + aOff);
      v8h hi8 = *(const v8h*)(aRow[mi] + k0 + aOff + 16);
      af[mi] = __builtin_shufflevector(lo, hi8, 0,1,2,3,4,5,6,7,8,9,10,11,12,13,14,15);
    }
    v16h bf[4];
#pragma unroll
    for (int ni = 0; ni < 4; ++ni) {
      v8h lo = *(const v8h*)(bRow[ni] + k0 + bOff);
      v8h hi8 = *(const v8h*)(bRow[ni] + k0 + bOff + 16);
      bf[ni] = __builtin_shufflevector(lo, hi8, 0,1,2,3,4,5,6,7,8,9,10,11,12,13,14,15);
    }

#pragma unroll
    for (int mi = 0; mi < 2; ++mi)
#pragma unroll
      for (int ni = 0; ni < 4; ++ni)
        acc[mi][ni] = __builtin_amdgcn_wmma_f32_16x16x32_f16(
            false, af[mi], false, bf[ni], (short)0, acc[mi][ni], false, false);
  }

  // C/D frag: VGPR v, lane<16 -> (M=v, N=lane); lane>=16 -> (M=v+8, N=lane-16)
#pragma unroll
  for (int mi = 0; mi < 2; ++mi) {
#pragma unroll
    for (int ni = 0; ni < 4; ++ni) {
      const int n = nWave + ni * 16 + l15;
      const float bv = bias[n];
      float* cp = C + (size_t)(mWave + mi * 16 + hi * 8) * N + n;
#pragma unroll
      for (int v = 0; v < 8; ++v) {
        cp[(size_t)v * N] = acc[mi][ni][v] + bv;
      }
    }
  }
}

// --------------------------------------------------------------------------
extern "C" void kernel_launch(void* const* d_in, const int* in_sizes, int n_in,
                              void* d_out, int out_size, void* d_ws, size_t ws_size,
                              hipStream_t stream) {
  (void)n_in; (void)out_size; (void)ws_size;
  const float* x      = (const float*)d_in[0];   // (B, I, 3)
  const float* weight = (const float*)d_in[1];   // (O, I)
  const float* action = (const float*)d_in[2];   // (O, I, 8)
  const float* e0     = (const float*)d_in[3];   // (I, 1)
  float* out = (float*)d_out;                    // (B, O, 3)

  const int I = in_sizes[3];                 // 512
  const int O = in_sizes[1] / I;             // 512
  const int M = in_sizes[0] / (3 * I);       // 4096 (batch)
  const int K = 3 * I;                       // 1536
  const int N = 3 * O;                       // 1536

  // workspace carve-out (256B aligned): A16 | Bt16 | biasC | bias  (~20.5 MB)
  char* base = (char*)d_ws;
  size_t off = 0;
  auto wsalloc = [&](size_t bytes) -> void* {
    void* p = base + off;
    off += (bytes + 255) & ~(size_t)255;
    return p;
  };
  _Float16* A16   = (_Float16*)wsalloc((size_t)M * K * sizeof(_Float16));
  _Float16* Bt16  = (_Float16*)wsalloc((size_t)N * K * sizeof(_Float16));
  float*    biasC = (float*)   wsalloc((size_t)I * N * sizeof(float));
  float*    bias  = (float*)   wsalloc((size_t)N * sizeof(float));

  { // 1) x -> f16 A matrix (already the right layout: (M, K) row-major)
    int n4 = (M * K) / 4;
    pga_xcast_kernel<<<dim3((n4 + 255) / 256), dim3(256), 0, stream>>>(x, A16, n4);
  }
  { // 2) per-(o,i) sandwich table -> Weff^T and bias contributions
    int nt = O * I;
    pga_build_weff_kernel<<<dim3((nt + 255) / 256), dim3(256), 0, stream>>>(
        weight, action, e0, Bt16, biasC, O, I);
  }
  // 3) deterministic, coalesced bias reduction
  pga_bias_reduce_kernel<<<dim3((N + 255) / 256), dim3(256), 0, stream>>>(biasC, bias, N, I);
  // 4) WMMA GEMM with fused bias epilogue (block tile 128x128)
  pga_gemm_kernel<<<dim3(M / 128, N / 128), dim3(256), 0, stream>>>(A16, Bt16, bias, out, M, N, K);
}
